// OneScalePredictor_75823352644380
// MI455X (gfx1250) — compile-verified, matrix operand order checked by reading, not run on hardware
//
#include <hip/hip_runtime.h>
#include <hip/hip_bf16.h>
#include <stdint.h>

// ---------------------------------------------------------------------------
// Types for CDNA5 (gfx1250) wave32 WMMA
// ---------------------------------------------------------------------------
typedef __attribute__((ext_vector_type(16))) __bf16 v16bf;
typedef __attribute__((ext_vector_type(8)))  float  v8f;
typedef __attribute__((ext_vector_type(4)))  unsigned int uint4v;

#define LOG2_E 1.4426950408889634f
#define C_CH   64
#define KTAPS  27
#define WAVES_PER_BLOCK 8
#define TM 4   // M-tiles (16 rows each) per wave: B fragments reused 4x

__device__ __forceinline__ unsigned short f2bf(float f) {
    unsigned u = __float_as_uint(f);
    unsigned r = 0x7FFFu + ((u >> 16) & 1u);   // round-to-nearest-even
    u += r;
    return (unsigned short)(u >> 16);
}

// ---------------------------------------------------------------------------
// Weight prep: f32 -> bf16, pre-swizzled into wave32 B-fragment layout.
// Fragment element e: v=e/2,h=e%2 -> K = kk*32 + (v<4 ? lo+2v+h
//   : 16+lo+2(v-4)+h), lo=(lane<16?0:8);  N = nt*16 + lane%16.
// ws layout: [5 sconv sets: 27*8 frags each][up_w1: 12 frags][pred_lw: 32].
// ---------------------------------------------------------------------------
__global__ void prep_weights_kernel(const float* __restrict__ w0,
                                    const float* __restrict__ w1,
                                    const float* __restrict__ w2,
                                    const float* __restrict__ w3,
                                    const float* __restrict__ w4,
                                    const float* __restrict__ upw1,
                                    const float* __restrict__ lw,
                                    unsigned short* __restrict__ frags) {
    const int per = KTAPS * 2 * 4 * 32;       // 6912 lane-fragments per set
    int t = blockIdx.x * blockDim.x + threadIdx.x;
    if (t < 5 * per) {
        int s = t / per, rr = t % per;
        int k = rr / 256, r2 = rr % 256;
        int kk = r2 / 128, nt = (r2 % 128) / 32, lane = r2 % 32;
        const float* W = (s == 0) ? w0 : (s == 1) ? w1 : (s == 2) ? w2
                       : (s == 3) ? w3 : w4;
        int lo = (lane < 16) ? 0 : 8;
        int N = nt * 16 + (lane & 15);
        unsigned short* o = frags + (size_t)s * (KTAPS * 8 * 512)
                          + ((size_t)((k * 2 + kk) * 4 + nt) * 32 + lane) * 16;
        #pragma unroll
        for (int e = 0; e < 16; e++) {
            int v = e >> 1, h = e & 1;
            int K = kk * 32 + ((v < 4) ? (lo + 2 * v + h)
                                       : (16 + lo + 2 * (v - 4) + h));
            o[e] = f2bf(W[(size_t)k * 4096 + K * 64 + N]);
        }
    } else if (t < 5 * per + 3 * 4 * 32) {    // up_w1: K padded 72 -> 96
        int t2 = t - 5 * per;
        int kk = t2 / 128, nt = (t2 % 128) / 32, lane = t2 % 32;
        int lo = (lane < 16) ? 0 : 8;
        int N = nt * 16 + (lane & 15);
        unsigned short* o = frags + (size_t)5 * (KTAPS * 8 * 512)
                          + ((size_t)(kk * 4 + nt) * 32 + lane) * 16;
        #pragma unroll
        for (int e = 0; e < 16; e++) {
            int v = e >> 1, h = e & 1;
            int K = kk * 32 + ((v < 4) ? (lo + 2 * v + h)
                                       : (16 + lo + 2 * (v - 4) + h));
            o[e] = f2bf((K < 72) ? upw1[(size_t)K * 64 + N] : 0.0f);
        }
    } else if (t < 5 * per + 3 * 4 * 32 + 2 * 16 * 32) {  // pred_lw 64x255->64x256
        int t3 = t - 5 * per - 3 * 4 * 32;
        int kk = t3 / 512, nt = (t3 % 512) / 32, lane = t3 % 32;
        int lo = (lane < 16) ? 0 : 8;
        int N = nt * 16 + (lane & 15);
        unsigned short* o = frags + (size_t)5 * (KTAPS * 8 * 512) + 3 * 4 * 32 * 16
                          + ((size_t)(kk * 16 + nt) * 32 + lane) * 16;
        #pragma unroll
        for (int e = 0; e < 16; e++) {
            int v = e >> 1, h = e & 1;
            int K = kk * 32 + ((v < 4) ? (lo + 2 * v + h)
                                       : (16 + lo + 2 * (v - 4) + h));
            o[e] = f2bf((N < 255) ? lw[(size_t)K * 255 + N] : 0.0f);
        }
    }
}

// ---------------------------------------------------------------------------
// f32 tensor -> bf16 tensor (packed b64 stores). Rows [totalF, totalAll)
// are zero-filled (zero row at index n used for masked gathers).
// ---------------------------------------------------------------------------
__global__ __launch_bounds__(256)
void cvt_bf16_kernel(const float* __restrict__ src,
                     unsigned short* __restrict__ dst,
                     int totalF, int totalAll) {
    int i = (blockIdx.x * 256 + threadIdx.x) * 4;
    if (i >= totalAll) return;
    uint2 pk;
    if (i < totalF) {
        const float4 v = *(const float4*)(src + i);
        pk.x = (unsigned)f2bf(v.x) | ((unsigned)f2bf(v.y) << 16);
        pk.y = (unsigned)f2bf(v.z) | ((unsigned)f2bf(v.w) << 16);
    } else {
        pk.x = 0u; pk.y = 0u;
    }
    *(uint2*)(dst + i) = pk;
}

// ---------------------------------------------------------------------------
// Sparse conv via WMMA, register-blocked TM=4 M-tiles per wave so each B
// fragment load is reused 4x (L2-traffic balance vs WMMA throughput).
// A gathered DIRECTLY from global bf16 (lane = A row M=lane%16; two
// 16B global_load_b128 per K-step). idx<0 remapped to zero row (index n).
// out = prelu(sum_k gather_k(G) @ W[k] + bias [+ res], alpha)  (f32 store)
// ---------------------------------------------------------------------------
__global__ __launch_bounds__(256)
void sconv_wmma_kernel(const unsigned short* __restrict__ G,  // (n+1) x 64 bf16
                       const unsigned short* __restrict__ Bfrag,
                       const float* __restrict__ bias,
                       const float* __restrict__ res,
                       const float* __restrict__ alphaPtr,
                       const int* __restrict__ nbr,
                       float* __restrict__ Fout, int n) {
    const int lane = threadIdx.x & 31;
    const int wib  = threadIdx.x >> 5;
    const int sg   = blockIdx.x * WAVES_PER_BLOCK + wib;  // super-tile group
    const int nTiles = n >> 4;
    const int tbase  = sg * TM;
    if (tbase >= nTiles) return;

    const float alpha = alphaPtr[0];
    const int col0 = lane & 15;
    const int lo   = (lane < 16) ? 0 : 8;

    v8f acc[TM][4];
    #pragma unroll
    for (int i = 0; i < TM; i++) {
        acc[i][0] = {}; acc[i][1] = {}; acc[i][2] = {}; acc[i][3] = {};
    }

    int mrow[TM];
    #pragma unroll
    for (int i = 0; i < TM; i++) {
        int mr = (tbase + i) * 16 + col0;       // A row this lane serves
        mrow[i] = (mr < n) ? mr : 0;            // clamp (acc discarded later)
    }

    for (int k = 0; k < KTAPS; k++) {
        union au { uint4v u[2]; v16bf b; };
        au a[TM][2];
        #pragma unroll
        for (int i = 0; i < TM; i++) {
            int idx  = nbr[(size_t)mrow[i] * KTAPS + k];
            int srow = (idx >= 0) ? idx : n;    // n = zero row
            const unsigned short* arow = G + (size_t)srow * C_CH;
            a[i][0].u[0] = *(const uint4v*)(arow + 0 * 32 + lo);
            a[i][0].u[1] = *(const uint4v*)(arow + 0 * 32 + lo + 16);
            a[i][1].u[0] = *(const uint4v*)(arow + 1 * 32 + lo);
            a[i][1].u[1] = *(const uint4v*)(arow + 1 * 32 + lo + 16);
        }

        const unsigned short* bk = Bfrag + (size_t)k * (8 * 512);
        #pragma unroll
        for (int nt = 0; nt < 4; nt++) {
            au b0, b1;
            const unsigned short* q0 = bk + ((size_t)(0 * 4 + nt) * 32 + lane) * 16;
            b0.u[0] = *(const uint4v*)(q0);
            b0.u[1] = *(const uint4v*)(q0 + 8);
            const unsigned short* q1 = bk + ((size_t)(1 * 4 + nt) * 32 + lane) * 16;
            b1.u[0] = *(const uint4v*)(q1);
            b1.u[1] = *(const uint4v*)(q1 + 8);

            #pragma unroll
            for (int i = 0; i < TM; i++) {
                acc[i][nt] = __builtin_amdgcn_wmma_f32_16x16x32_bf16(
                    false, a[i][0].b, false, b0.b, (short)0, acc[i][nt], false, false);
                acc[i][nt] = __builtin_amdgcn_wmma_f32_16x16x32_bf16(
                    false, a[i][1].b, false, b1.b, (short)0, acc[i][nt], false, false);
            }
        }
    }

    const int roff = (lane < 16) ? 0 : 8;
    #pragma unroll
    for (int nt = 0; nt < 4; nt++) {
        int col = nt * 16 + col0;
        float bcol = bias[col];
        #pragma unroll
        for (int i = 0; i < TM; i++) {
            int rbase = (tbase + i) * 16 + roff;
            #pragma unroll
            for (int v = 0; v < 8; v++) {
                int row = rbase + v;
                if (row < n) {
                    float x = acc[i][nt][v] + bcol;
                    if (res) x += res[(size_t)row * C_CH + col];
                    x = (x >= 0.0f) ? x : alpha * x;
                    Fout[(size_t)row * C_CH + col] = x;
                }
            }
        }
    }
}

// ---------------------------------------------------------------------------
// up1: u = prelu([h | cur_bin | 0pad] @ up_w1 + b, a). K=96 (3 K-steps).
// h is read from its bf16 copy; LDS assembles the mixed-source A tile.
// ---------------------------------------------------------------------------
__global__ __launch_bounds__(256)
void up1_wmma_kernel(const unsigned short* __restrict__ Hbf,   // (n+1) x 64 bf16
                     const float* __restrict__ bin,
                     const unsigned short* __restrict__ Bfrag,
                     const float* __restrict__ bias,
                     const float* __restrict__ alphaPtr,
                     float* __restrict__ Uout, int n) {
    __shared__ unsigned short smem[WAVES_PER_BLOCK * 16 * 96];
    const int lane = threadIdx.x & 31;
    const int wib  = threadIdx.x >> 5;
    const int tile = blockIdx.x * WAVES_PER_BLOCK + wib;
    const int nTiles = n >> 4;
    if (tile >= nTiles) return;

    unsigned short* lds = smem + wib * (16 * 96);
    const float alpha = alphaPtr[0];

    const int r  = lane >> 1;
    const int hs = lane & 1;
    const int grow = tile * 16 + r;
    unsigned short* drow = lds + r * 96 + hs * 48;
    const unsigned short* hrow = Hbf + (size_t)grow * C_CH;
    if (hs == 0) {                               // h[0..47] (bf16 copy)
        #pragma unroll
        for (int q = 0; q < 48; q++) drow[q] = hrow[q];
    } else {                                      // h[48..63] | bin[0..7] | 0
        #pragma unroll
        for (int q = 0; q < 16; q++) drow[q] = hrow[48 + q];
        const float* bsrc = bin + (size_t)grow * 8;
        #pragma unroll
        for (int q = 0; q < 8; q++) drow[16 + q] = f2bf(bsrc[q]);
        #pragma unroll
        for (int q = 0; q < 24; q++) drow[24 + q] = 0;
    }

    v8f acc[4];
    acc[0] = {}; acc[1] = {}; acc[2] = {}; acc[3] = {};
    const int lo = (lane < 16) ? 0 : 8;
    const int aRow = (lane & 15) * 96;

    #pragma unroll
    for (int kk = 0; kk < 3; kk++) {
        union { uint4v u[2]; v16bf b; } a;
        const unsigned short* p = lds + aRow + kk * 32 + lo;
        a.u[0] = *(const uint4v*)(p);
        a.u[1] = *(const uint4v*)(p + 16);
        #pragma unroll
        for (int nt = 0; nt < 4; nt++) {
            union { uint4v u[2]; v16bf b; } bf;
            const unsigned short* q = Bfrag + ((size_t)(kk * 4 + nt) * 32 + lane) * 16;
            bf.u[0] = *(const uint4v*)(q);
            bf.u[1] = *(const uint4v*)(q + 8);
            acc[nt] = __builtin_amdgcn_wmma_f32_16x16x32_bf16(
                false, a.b, false, bf.b, (short)0, acc[nt], false, false);
        }
    }

    const int col0  = lane & 15;
    const int rbase = tile * 16 + ((lane < 16) ? 0 : 8);
    #pragma unroll
    for (int nt = 0; nt < 4; nt++) {
        int col = nt * 16 + col0;
        float bcol = bias[col];
        #pragma unroll
        for (int v = 0; v < 8; v++) {
            int row = rbase + v;
            float x = acc[nt][v] + bcol;
            x = (x >= 0.0f) ? x : alpha * x;
            Uout[(size_t)row * C_CH + col] = x;
        }
    }
}

// ---------------------------------------------------------------------------
// WMMA cross-entropy head: wave per 16-row tile. logits(16x256) = p @ lw
// (2 K-steps x 16 N-tiles), bias folded in; col 255 masked with -3e38 bias.
// log-softmax reductions via shfl_xor within each 16-lane half
// (C/D layout: row = v + 8*(lane>=16), col = nt*16 + lane%16).
// ---------------------------------------------------------------------------
__global__ __launch_bounds__(256)
void ce_wmma_kernel(const unsigned short* __restrict__ Pbf,    // n x 64 bf16
                    const unsigned short* __restrict__ Bfrag,  // 32 frags
                    const float* __restrict__ lb,
                    const float* __restrict__ bin,
                    const int* __restrict__ batch_ids,
                    const int* __restrict__ pnum,
                    float* __restrict__ ce, int n) {
    const int lane = threadIdx.x & 31;
    const int wib  = threadIdx.x >> 5;
    const int tile = blockIdx.x * WAVES_PER_BLOCK + wib;
    const int nTiles = n >> 4;
    if (tile >= nTiles) return;

    const int col0 = lane & 15;
    const int lo   = (lane < 16) ? 0 : 8;

    float lbv[16];
    #pragma unroll
    for (int nt = 0; nt < 16; nt++) {
        int col = nt * 16 + col0;
        lbv[nt] = (col < 255) ? lb[col] : -3.0e38f;   // mask padded col 255
    }

    // A fragments: p rows, direct from bf16 global
    const unsigned short* arow = Pbf + (size_t)(tile * 16 + col0) * C_CH;
    union { uint4v u[2]; v16bf b; } a0, a1;
    a0.u[0] = *(const uint4v*)(arow + 0 * 32 + lo);
    a0.u[1] = *(const uint4v*)(arow + 0 * 32 + lo + 16);
    a1.u[0] = *(const uint4v*)(arow + 1 * 32 + lo);
    a1.u[1] = *(const uint4v*)(arow + 1 * 32 + lo + 16);

    v8f acc[16];
    #pragma unroll
    for (int nt = 0; nt < 16; nt++) acc[nt] = {};

    #pragma unroll
    for (int nt = 0; nt < 16; nt++) {
        union { uint4v u[2]; v16bf b; } b0, b1;
        const unsigned short* q0 = Bfrag + ((size_t)(0 * 16 + nt) * 32 + lane) * 16;
        b0.u[0] = *(const uint4v*)(q0);
        b0.u[1] = *(const uint4v*)(q0 + 8);
        const unsigned short* q1 = Bfrag + ((size_t)(1 * 16 + nt) * 32 + lane) * 16;
        b1.u[0] = *(const uint4v*)(q1);
        b1.u[1] = *(const uint4v*)(q1 + 8);
        acc[nt] = __builtin_amdgcn_wmma_f32_16x16x32_bf16(
            false, a0.b, false, b0.b, (short)0, acc[nt], false, false);
        acc[nt] = __builtin_amdgcn_wmma_f32_16x16x32_bf16(
            false, a1.b, false, b1.b, (short)0, acc[nt], false, false);
    }

    const int rbase = tile * 16 + ((lane < 16) ? 0 : 8);
    #pragma unroll
    for (int v = 0; v < 8; v++) {
        int row = rbase + v;
        // target octant code for this row (all 16 lanes of the half agree)
        const float* b = bin + (size_t)row * 8;
        int tgt = -1;
        #pragma unroll
        for (int j = 0; j < 8; j++) tgt += (b[j] != 0.0f) ? (1 << j) : 0;

        float mx = -3.4e38f;
        #pragma unroll
        for (int nt = 0; nt < 16; nt++) mx = fmaxf(mx, acc[nt][v] + lbv[nt]);
        #pragma unroll
        for (int off = 8; off >= 1; off >>= 1) mx = fmaxf(mx, __shfl_xor(mx, off));

        float se = 0.0f;
        #pragma unroll
        for (int nt = 0; nt < 16; nt++) se += __expf(acc[nt][v] + lbv[nt] - mx);
        #pragma unroll
        for (int off = 8; off >= 1; off >>= 1) se += __shfl_xor(se, off);

        float lt = -3.4e38f;
        #pragma unroll
        for (int nt = 0; nt < 16; nt++) {
            int col = nt * 16 + col0;
            if (col == tgt) lt = acc[nt][v] + lbv[nt];
        }
        #pragma unroll
        for (int off = 8; off >= 1; off >>= 1) lt = fmaxf(lt, __shfl_xor(lt, off));

        if ((lane & 15) == 0) {
            float logp  = lt - mx - __logf(se);
            float denom = (float)pnum[batch_ids[row]];
            ce[row] = (-logp) / denom;
        }
    }
}

__global__ __launch_bounds__(256)
void reduce_kernel(const float* __restrict__ ce, int n,
                   float* __restrict__ outLoss, float scale) {
    __shared__ float sm[256];
    float s = 0.0f;
    for (int i = threadIdx.x; i < n; i += 256) s += ce[i];
    sm[threadIdx.x] = s;
    __syncthreads();
    for (int off = 128; off >= 1; off >>= 1) {
        if ((int)threadIdx.x < off) sm[threadIdx.x] += sm[threadIdx.x + off];
        __syncthreads();
    }
    if (threadIdx.x == 0) *outLoss = sm[0] * scale;
}

// ---------------------------------------------------------------------------
// Final: out[s] = u3[rows[s]] @ up_w2[:, slots[s]*64 : +64] + up_b2[slice].
// Wave per selected row; lane j covers cols j and j+32.
// ---------------------------------------------------------------------------
__global__ __launch_bounds__(256)
void out_kernel(const float* __restrict__ U, const float* __restrict__ W2,
                const float* __restrict__ b2, const int* __restrict__ rows,
                const int* __restrict__ slots, float* __restrict__ out, int m) {
    int wid  = (blockIdx.x * blockDim.x + threadIdx.x) >> 5;
    int lane = threadIdx.x & 31;
    if (wid >= m) return;
    int r = rows[wid], sl = slots[wid];
    const float* u = U + (size_t)r * C_CH;
    const float* w = W2 + (size_t)sl * 64;
    float a0 = b2[sl * 64 + lane];
    float a1 = b2[sl * 64 + lane + 32];
    for (int c = 0; c < C_CH; c++) {
        float uc = u[c];
        a0 += uc * w[(size_t)c * 512 + lane];
        a1 += uc * w[(size_t)c * 512 + lane + 32];
    }
    out[(size_t)wid * 64 + lane]      = a0;
    out[(size_t)wid * 64 + lane + 32] = a1;
}

// ---------------------------------------------------------------------------
// Host-side orchestration
// ---------------------------------------------------------------------------
static inline size_t align256(size_t x) { return (x + 255) & ~(size_t)255; }

extern "C" void kernel_launch(void* const* d_in, const int* in_sizes, int n_in,
                              void* d_out, int out_size, void* d_ws, size_t ws_size,
                              hipStream_t stream) {
    (void)n_in; (void)ws_size;
    const float* cur_rec_F = (const float*)d_in[0];
    const float* cur_bin   = (const float*)d_in[1];
    const float* dec_w1    = (const float*)d_in[2];
    const float* dec_b1    = (const float*)d_in[3];
    const float* dec_a1    = (const float*)d_in[4];
    const float* dec_w2    = (const float*)d_in[5];
    const float* dec_b2    = (const float*)d_in[6];
    const float* dec_a2    = (const float*)d_in[7];
    const float* pred_cw   = (const float*)d_in[8];
    const float* pred_cb   = (const float*)d_in[9];
    const float* pred_ca   = (const float*)d_in[10];
    const float* pred_lw   = (const float*)d_in[11];
    const float* pred_lb   = (const float*)d_in[12];
    const float* up_w1     = (const float*)d_in[13];
    const float* up_b1     = (const float*)d_in[14];
    const float* up_a      = (const float*)d_in[15];
    const float* ubk_w1    = (const float*)d_in[16];
    const float* ubk_b1    = (const float*)d_in[17];
    const float* ubk_a1    = (const float*)d_in[18];
    const float* ubk_w2    = (const float*)d_in[19];
    const float* ubk_b2    = (const float*)d_in[20];
    const float* ubk_a2    = (const float*)d_in[21];
    const float* up_w2     = (const float*)d_in[22];
    const float* up_b2     = (const float*)d_in[23];
    const int*   pnum      = (const int*)d_in[24];
    const int*   batch_ids = (const int*)d_in[25];
    const int*   nbr_idx   = (const int*)d_in[27];
    const int*   sel_rows  = (const int*)d_in[28];
    const int*   sel_slots = (const int*)d_in[29];

    const int n  = in_sizes[0] / C_CH;       // 131072
    const int bs = in_sizes[24];             // batch size
    const int m  = in_sizes[28];             // # selected rows

    // fragment region sizes (ushorts)
    const size_t SET_US  = (size_t)KTAPS * 8 * 512;        // per sconv set
    const size_t UP1_US  = (size_t)3 * 4 * 32 * 16;        // 6144
    const size_t LW_US   = (size_t)2 * 16 * 32 * 16;       // 16384
    unsigned short* frags  = (unsigned short*)d_ws;
    unsigned short* upfrag = frags + 5 * SET_US;
    unsigned short* lwfrag = upfrag + UP1_US;

    size_t off = align256((5 * SET_US + UP1_US + LW_US) * 2);
    float* ce = (float*)((char*)d_ws + off);
    off += align256((size_t)n * 4);
    unsigned short* G0 = (unsigned short*)((char*)d_ws + off);
    off += align256((size_t)(n + 1) * C_CH * 2);
    unsigned short* G1 = (unsigned short*)((char*)d_ws + off);
    off += align256((size_t)(n + 1) * C_CH * 2);
    float* T0 = (float*)((char*)d_ws + off);
    float* T1 = T0 + (size_t)n * C_CH;
    float* T2 = T1 + (size_t)n * C_CH;

    // 1) weight prep
    {
        int total = 5 * (KTAPS * 2 * 4 * 32) + 3 * 4 * 32 + 2 * 16 * 32;
        prep_weights_kernel<<<(total + 255) / 256, 256, 0, stream>>>(
            dec_w1, dec_w2, pred_cw, ubk_w1, ubk_w2, up_w1, pred_lw, frags);
    }

    const int nTiles = n / 16;
    const int gconv  = (nTiles + WAVES_PER_BLOCK * TM - 1) / (WAVES_PER_BLOCK * TM);
    const int gtile  = (nTiles + WAVES_PER_BLOCK - 1) / WAVES_PER_BLOCK;
    const int totalF   = n * C_CH;
    const int totalAll = (n + 1) * C_CH;
    const int gcvt     = (totalAll / 4 + 255) / 256;

    // 2) F -> bf16 (with zero row)
    cvt_bf16_kernel<<<gcvt, 256, 0, stream>>>(cur_rec_F, G0, totalF, totalAll);
    // 3) h1 = prelu(sconv(F, dec_w1)+b1, a1)
    sconv_wmma_kernel<<<gconv, 256, 0, stream>>>(
        G0, frags + 0 * SET_US, dec_b1, nullptr, dec_a1, nbr_idx, T0, n);
    cvt_bf16_kernel<<<gcvt, 256, 0, stream>>>(T0, G1, totalF, totalAll);
    // 4) h = prelu(sconv(h1, dec_w2)+b2 + F, a2)
    sconv_wmma_kernel<<<gconv, 256, 0, stream>>>(
        G1, frags + 1 * SET_US, dec_b2, cur_rec_F, dec_a2, nbr_idx, T1, n);
    cvt_bf16_kernel<<<gcvt, 256, 0, stream>>>(T1, G0, totalF, totalAll);
    // 5) p = prelu(sconv(h, pred_cw)+cb, ca)
    sconv_wmma_kernel<<<gconv, 256, 0, stream>>>(
        G0, frags + 2 * SET_US, pred_cb, nullptr, pred_ca, nbr_idx, T2, n);
    cvt_bf16_kernel<<<gcvt, 256, 0, stream>>>(T2, G1, totalF, totalAll);
    // 6) WMMA logits + fused log-softmax cross entropy
    ce_wmma_kernel<<<gtile, 256, 0, stream>>>(
        G1, lwfrag, pred_lb, cur_bin, batch_ids, pnum, ce, n);
    // 7) scalar loss -> last element of d_out
    reduce_kernel<<<1, 256, 0, stream>>>(
        ce, n, (float*)d_out + (out_size - 1), LOG2_E / (float)bs);
    // 8) u1 = prelu([h|bin] @ up_w1 + b, a)   (h bf16 still in G0)
    up1_wmma_kernel<<<gtile, 256, 0, stream>>>(
        G0, cur_bin, upfrag, up_b1, up_a, T0, n);
    cvt_bf16_kernel<<<gcvt, 256, 0, stream>>>(T0, G1, totalF, totalAll);
    // 9) u2 = prelu(sconv(u1, ubk_w1)+b1, a1)
    sconv_wmma_kernel<<<gconv, 256, 0, stream>>>(
        G1, frags + 3 * SET_US, ubk_b1, nullptr, ubk_a1, nbr_idx, T2, n);
    cvt_bf16_kernel<<<gcvt, 256, 0, stream>>>(T2, G0, totalF, totalAll);
    // 10) u3 = prelu(sconv(u2, ubk_w2)+b2 + u1, a2)
    sconv_wmma_kernel<<<gconv, 256, 0, stream>>>(
        G0, frags + 4 * SET_US, ubk_b2, T0, ubk_a2, nbr_idx, T1, n);
    // 11) selected output rows
    out_kernel<<<(m * 32 + 255) / 256, 256, 0, stream>>>(
        T1, up_w2, up_b2, sel_rows, sel_slots, (float*)d_out, m);
}
